// LIFModule_7189775253505
// MI455X (gfx1250) — matrix operand order repaired and checked by reading, not compiled, not run on hardware
//
#include <hip/hip_runtime.h>

// ---------------------------------------------------------------------------
// LIF spiking block for MI455X (gfx1250, wave32, WMMA).
// Memory-bound (~10 GFLOP vs hundreds of MB @ 23.3 TB/s): elementwise stages
// (GN apply / leaky / LIF) are fused into producers/consumers; the 48x48
// channel-mix convs run on v_wmma_f32_16x16x32_f16 with LDS-staged fragments
// (coalesced b128 global loads, contiguous ds_load_b128 fragment reads,
// no divergent guards around the WMMAs).
// ---------------------------------------------------------------------------

typedef __attribute__((ext_vector_type(16))) _Float16 v16h;
typedef __attribute__((ext_vector_type(8)))  float    v8f;

#define Bsz    8
#define DIM    48
#define Hdim   256
#define Wdim   256
#define PLANE  (Hdim * Wdim)          // 65536
#define GROUPS 24
#define EPS    1e-5f
#define SLOPE  0.1f
#define LIF_T  4

#define BSTRIDE 72                    // LDS B-tile row stride in halves (144 B)

__device__ __forceinline__ float leakyf(float v) {
    return v >= 0.f ? v : SLOPE * v;
}

// ---------------------------------------------------------------------------
// 1x1 conv (48x48 channel mix) over N = B*H*W positions via WMMA f16.
//   out[o,n] = sum_c w[o,c] * in[c,n] + bias[o]
// Block: 256 threads = 8 waves; 128 consecutive spatial positions per block
// (16 per wave). K = 48 padded to 64 (2 WMMA K-steps), M = 3 tiles of 16.
// Input tile staged to LDS as f16 with coalesced float4 global loads;
// A/B fragments are contiguous 16B runs in LDS (ds_load_b128).
// IN_GN:     apply group-norm to the input during staging (no leaky; conv3).
// OUT_STATS: accumulate per-(b,group) sum/sumsq of (conv + bias) output.
// ---------------------------------------------------------------------------
template<int IN_GN, int OUT_STATS>
__global__ __launch_bounds__(256)
void conv1x1_wmma(const float* __restrict__ in,
                  const float* __restrict__ w,      // [48][48]
                  const float* __restrict__ bias,   // [48]
                  float* __restrict__ out,
                  const float* __restrict__ inMean, // [B*24] (if IN_GN)
                  const float* __restrict__ inRsig, // [B*24]
                  const float* __restrict__ inGamma,// [48]
                  const float* __restrict__ inBeta, // [48]
                  float* __restrict__ outSum,       // [B*24] (if OUT_STATS)
                  float* __restrict__ outSq)        // [B*24]
{
    __shared__ _Float16 wlds[DIM * 64];        // weights [o][k], K padded
    __shared__ _Float16 blds[128 * BSTRIDE];   // input tile [n_local][k]
    __shared__ float    statLds[2 * DIM];      // [0..47]=sum, [48..95]=sumsq

    const int tid  = threadIdx.x;
    const int b    = blockIdx.x >> 9;          // 512 blocks per batch
    const int pos0 = (blockIdx.x & 511) << 7;  // 128 positions per block
    const size_t planeBase = (size_t)b * DIM * PLANE;

    // Stage weights into LDS as f16, zero-padded in K.
    for (int i = tid; i < DIM * 64; i += 256) {
        const int o = i >> 6, k = i & 63;
        wlds[i] = (k < DIM) ? (_Float16)w[o * DIM + k] : (_Float16)0.f;
    }
    // Zero the K-pad region of the B tile (k = 48..63).
    for (int i = tid; i < 128 * 16; i += 256) {
        const int n = i >> 4, k = DIM + (i & 15);
        blds[n * BSTRIDE + k] = (_Float16)0.f;
    }
    if (OUT_STATS) {
        for (int i = tid; i < 2 * DIM; i += 256) statLds[i] = 0.f;
    }

    // Stage input tile: 48 channels x 128 positions, coalesced float4 loads.
    for (int i = tid; i < DIM * 32; i += 256) {        // 1536 float4s
        const int c = i >> 5, q = i & 31;
        float4 v = *reinterpret_cast<const float4*>(
            in + planeBase + (size_t)c * PLANE + pos0 + (q << 2));
        if (IN_GN) {
            const int g = c >> 1;
            const float sc = inRsig[b * GROUPS + g] * inGamma[c];
            const float sh = inBeta[c] - inMean[b * GROUPS + g] * sc;
            v.x = v.x * sc + sh; v.y = v.y * sc + sh;
            v.z = v.z * sc + sh; v.w = v.w * sc + sh;
        }
        const int n = q << 2;
        blds[(n + 0) * BSTRIDE + c] = (_Float16)v.x;
        blds[(n + 1) * BSTRIDE + c] = (_Float16)v.y;
        blds[(n + 2) * BSTRIDE + c] = (_Float16)v.z;
        blds[(n + 3) * BSTRIDE + c] = (_Float16)v.w;
    }
    __syncthreads();

    const int lane    = tid & 31;
    const int wv      = tid >> 5;              // wave id 0..7
    const int halfSel = lane >> 4;             // lane group (0 / 1)
    const int ln      = (wv << 4) + (lane & 15);   // local position 0..127
    const int n       = pos0 + ln;             // position within plane

    v8f acc[3] = {};
#pragma unroll
    for (int ks = 0; ks < 2; ++ks) {
        // B fragment (ISA 16-bit B layout): lane holds N = lane%16,
        // two contiguous 16B runs at k = ks*32 + halfSel*8 (+16).
        v16h bfrag;
        const _Float16* bp = &blds[ln * BSTRIDE + ks * 32 + halfSel * 8];
#pragma unroll
        for (int e = 0; e < 8; ++e) {
            bfrag[e]     = bp[e];
            bfrag[8 + e] = bp[16 + e];
        }

#pragma unroll
        for (int mt = 0; mt < 3; ++mt) {
            // A fragment (ISA 16-bit A layout): lane holds M = lane%16,
            // two contiguous 16B runs at k = ks*32 + halfSel*8 (+16).
            v16h afrag;
            const _Float16* ap =
                &wlds[(mt * 16 + (lane & 15)) * 64 + ks * 32 + halfSel * 8];
#pragma unroll
            for (int e = 0; e < 8; ++e) {
                afrag[e]     = ap[e];
                afrag[8 + e] = ap[16 + e];
            }
            acc[mt] = __builtin_amdgcn_wmma_f32_16x16x32_f16(
                false, afrag, false, bfrag, (short)0, acc[mt], false, false);
        }
    }

    // Write out (+bias), fuse GN output statistics.
#pragma unroll
    for (int mt = 0; mt < 3; ++mt) {
#pragma unroll
        for (int r = 0; r < 8; ++r) {
            const int o = mt * 16 + halfSel * 8 + r;  // C/D layout
            const float val = acc[mt][r] + bias[o];
            out[planeBase + (size_t)o * PLANE + n] = val;
            if (OUT_STATS) {
                atomicAdd(&statLds[o], val);
                atomicAdd(&statLds[DIM + o], val * val);
            }
        }
    }
    if (OUT_STATS) {
        __syncthreads();
        if (tid < DIM) {
            atomicAdd(&outSum[b * GROUPS + (tid >> 1)], statLds[tid]);
            atomicAdd(&outSq [b * GROUPS + (tid >> 1)], statLds[DIM + tid]);
        }
    }
}

// ---------------------------------------------------------------------------
// (sum, sumsq) -> (mean, rsqrt(var+eps)), in place. 192 = B*GROUPS threads.
// ---------------------------------------------------------------------------
__global__ void reduce_stats(float* sum, float* sq, float invCnt)
{
    const int i = threadIdx.x;
    const float mean = sum[i] * invCnt;
    const float var  = sq[i] * invCnt - mean * mean;
    sum[i] = mean;
    sq[i]  = rsqrtf(var + EPS);
}

// ---------------------------------------------------------------------------
// Depthwise 3x3 (SAME). Input = leaky(GN1(y1)) applied at load via LDS halo
// tile. Fuses GN2 output stats. Block = 256 threads -> 16x16 output tile.
// ---------------------------------------------------------------------------
__global__ __launch_bounds__(256)
void dwconv3x3(const float* __restrict__ y1,
               const float* __restrict__ wdw,  // [48][1][3][3]
               const float* __restrict__ bdw,  // [48]
               const float* __restrict__ mean1, const float* __restrict__ rsig1,
               const float* __restrict__ g1,    const float* __restrict__ be1,
               float* __restrict__ y2,
               float* __restrict__ sum2, float* __restrict__ sq2)
{
    __shared__ float tile[18 * 18];
    __shared__ float wk[10];
    __shared__ float red[2];

    const int tid  = threadIdx.x;
    const int bidx = blockIdx.x;                 // (b*48+c)*256 + tH*16 + tW
    const int tW = bidx & 15;
    const int tH = (bidx >> 4) & 15;
    const int c  = (bidx >> 8) % DIM;
    const int b  = (bidx >> 8) / DIM;
    const size_t plane = ((size_t)b * DIM + c) * PLANE;

    const float mu = mean1[b * GROUPS + (c >> 1)];
    const float rs = rsig1[b * GROUPS + (c >> 1)];
    const float ga = g1[c], bb = be1[c];

    if (tid < 9)  wk[tid] = wdw[c * 9 + tid];
    if (tid == 9) wk[9]   = bdw[c];
    if (tid < 2)  red[tid] = 0.f;

    for (int i = tid; i < 18 * 18; i += 256) {
        const int ly = i / 18, lx = i % 18;
        const int hy = tH * 16 + ly - 1, hx = tW * 16 + lx - 1;
        float v = 0.f;
        if (hy >= 0 && hy < Hdim && hx >= 0 && hx < Wdim) {
            const float raw = y1[plane + hy * Wdim + hx];
            v = leakyf((raw - mu) * rs * ga + bb);
        }
        tile[i] = v;
    }
    __syncthreads();

    const int ty = tid >> 4, tx = tid & 15;
    float s = wk[9];
#pragma unroll
    for (int di = 0; di < 3; ++di)
#pragma unroll
        for (int dj = 0; dj < 3; ++dj)
            s += tile[(ty + di) * 18 + tx + dj] * wk[di * 3 + dj];

    y2[plane + (tH * 16 + ty) * Wdim + tW * 16 + tx] = s;

    atomicAdd(&red[0], s);
    atomicAdd(&red[1], s * s);
    __syncthreads();
    if (tid == 0) {
        atomicAdd(&sum2[b * GROUPS + (c >> 1)], red[0]);
        atomicAdd(&sq2 [b * GROUPS + (c >> 1)], red[1]);
    }
}

// ---------------------------------------------------------------------------
// LIF over 4 chunks of axis H (alongW=0) or axis W (alongW=1).
// Input = leaky(GN2(y2)) applied on the fly. mem=mem*tau+x; s=(mem-vth>0);
// mem*=(1-s). Spikes (0/1) written to the same layout.
// ---------------------------------------------------------------------------
__global__ __launch_bounds__(256)
void lif_kernel(const float* __restrict__ y2,
                const float* __restrict__ mean2, const float* __restrict__ rsig2,
                const float* __restrict__ g2,    const float* __restrict__ be2,
                const float* __restrict__ tauP,  const float* __restrict__ vthP,
                float* __restrict__ spikes, int alongW)
{
    const int gid = blockIdx.x * 256 + threadIdx.x;
    const int bc    = gid >> 14;           // 16384 threads per (b,c) plane
    const int local = gid & 16383;
    const int c = bc % DIM, b = bc / DIM;
    const size_t plane = (size_t)bc * PLANE;

    int off0, stride;
    if (!alongW) {                         // chunks of 64 rows
        const int wpos = local & 255, hh = local >> 8;
        off0 = hh * Wdim + wpos; stride = 64 * Wdim;
    } else {                               // chunks of 64 columns
        const int ww = local & 63, h = local >> 6;
        off0 = h * Wdim + ww; stride = 64;
    }

    const float mu = mean2[b * GROUPS + (c >> 1)];
    const float rs = rsig2[b * GROUPS + (c >> 1)];
    const float ga = g2[c], bb = be2[c];
    const float tau = tauP[0], vth = vthP[0];

    float mem = 0.f;
#pragma unroll
    for (int t = 0; t < LIF_T; ++t) {
        const size_t idx = plane + off0 + (size_t)t * stride;
        const float hv = leakyf((y2[idx] - mu) * rs * ga + bb);
        mem = mem * tau + hv;
        const float s = (mem - vth) > 0.f ? 1.f : 0.f;
        spikes[idx] = s;
        mem *= (1.f - s);
    }
}

// ---------------------------------------------------------------------------
// s_pre = leaky(GN4(y4)) + leaky(GN5(y5)); fuse GN3 stats of s_pre.
// Block handles 4096 consecutive elements of one (b,c) plane (float4 I/O).
// ---------------------------------------------------------------------------
__global__ __launch_bounds__(256)
void add_gn(const float* __restrict__ y4, const float* __restrict__ y5,
            const float* __restrict__ m4, const float* __restrict__ r4,
            const float* __restrict__ g4, const float* __restrict__ be4,
            const float* __restrict__ m5, const float* __restrict__ r5,
            const float* __restrict__ g5, const float* __restrict__ be5,
            float* __restrict__ spre,
            float* __restrict__ sum3, float* __restrict__ sq3)
{
    __shared__ float red[2];
    const int tid  = threadIdx.x;
    const int bidx = blockIdx.x;                // (b*48+c)*16 + chunk
    const int chunk = bidx & 15;
    const int c = (bidx >> 4) % DIM;
    const int b = (bidx >> 4) / DIM;
    const size_t base = ((size_t)b * DIM + c) * PLANE + chunk * 4096;
    const int g = b * GROUPS + (c >> 1);

    const float sc4 = r4[g] * g4[c], sh4 = be4[c] - m4[g] * r4[g] * g4[c];
    const float sc5 = r5[g] * g5[c], sh5 = be5[c] - m5[g] * r5[g] * g5[c];

    if (tid < 2) red[tid] = 0.f;
    __syncthreads();

    const float4* a4 = reinterpret_cast<const float4*>(y4 + base);
    const float4* d4 = reinterpret_cast<const float4*>(y5 + base);
    float4*       s4 = reinterpret_cast<float4*>(spre + base);

    float ls = 0.f, lq = 0.f;
    for (int j = tid; j < 1024; j += 256) {
        const float4 a = a4[j];
        const float4 d = d4[j];
        float4 v;
        v.x = leakyf(a.x * sc4 + sh4) + leakyf(d.x * sc5 + sh5);
        v.y = leakyf(a.y * sc4 + sh4) + leakyf(d.y * sc5 + sh5);
        v.z = leakyf(a.z * sc4 + sh4) + leakyf(d.z * sc5 + sh5);
        v.w = leakyf(a.w * sc4 + sh4) + leakyf(d.w * sc5 + sh5);
        s4[j] = v;
        ls += v.x + v.y + v.z + v.w;
        lq += v.x * v.x + v.y * v.y + v.z * v.z + v.w * v.w;
    }
    atomicAdd(&red[0], ls);
    atomicAdd(&red[1], lq);
    __syncthreads();
    if (tid == 0) {
        atomicAdd(&sum3[g], red[0]);
        atomicAdd(&sq3 [g], red[1]);
    }
}

// ---------------------------------------------------------------------------
extern "C" void kernel_launch(void* const* d_in, const int* in_sizes, int n_in,
                              void* d_out, int out_size, void* d_ws, size_t ws_size,
                              hipStream_t stream)
{
    const float* x   = (const float*)d_in[0];
    const float* w1  = (const float*)d_in[1];
    const float* b1  = (const float*)d_in[2];
    const float* wdw = (const float*)d_in[3];
    const float* bdw = (const float*)d_in[4];
    const float* w21 = (const float*)d_in[5];
    const float* b21 = (const float*)d_in[6];
    const float* w22 = (const float*)d_in[7];
    const float* b22 = (const float*)d_in[8];
    const float* w3  = (const float*)d_in[9];
    const float* b3  = (const float*)d_in[10];
    const float* g1  = (const float*)d_in[11];
    const float* be1 = (const float*)d_in[12];
    const float* g2  = (const float*)d_in[13];
    const float* be2 = (const float*)d_in[14];
    const float* g3  = (const float*)d_in[15];
    const float* be3 = (const float*)d_in[16];
    const float* g4  = (const float*)d_in[17];
    const float* be4 = (const float*)d_in[18];
    const float* g5  = (const float*)d_in[19];
    const float* be5 = (const float*)d_in[20];
    const float* tau1 = (const float*)d_in[21];
    const float* vth1 = (const float*)d_in[22];
    const float* tau2 = (const float*)d_in[23];
    const float* vth2 = (const float*)d_in[24];

    char* ws = (char*)d_ws;
    float* stats = (float*)ws;                       // 5 stages x 384 floats
    const size_t SZ = (size_t)Bsz * DIM * PLANE;     // elements per tensor
    float* bufA = (float*)(ws + 8192);
    float* bufB = bufA + SZ;
    float* bufC = bufB + SZ;
    float* bufD = bufC + SZ;

    float* sum1 = stats;        float* sq1 = stats + 192;
    float* sum2 = stats + 384;  float* sq2 = stats + 576;
    float* sum3 = stats + 768;  float* sq3 = stats + 960;
    float* sum4 = stats + 1152; float* sq4 = stats + 1344;
    float* sum5 = stats + 1536; float* sq5 = stats + 1728;

    hipMemsetAsync(stats, 0, 5 * 384 * sizeof(float), stream);

    const float invCnt = 1.f / ((float)(DIM / GROUPS) * PLANE);   // per (b,group)
    const dim3 blk(256);

    // y1 = conv1x1(x, w1, b1); stats1
    conv1x1_wmma<0, 1><<<4096, blk, 0, stream>>>(
        x, w1, b1, bufA, nullptr, nullptr, nullptr, nullptr, sum1, sq1);
    reduce_stats<<<1, 192, 0, stream>>>(sum1, sq1, invCnt);

    // y2 = dwconv3x3(leaky(gn1(y1))); stats2
    dwconv3x3<<<Bsz * DIM * 256, blk, 0, stream>>>(
        bufA, wdw, bdw, sum1, sq1, g1, be1, bufB, sum2, sq2);
    reduce_stats<<<1, 192, 0, stream>>>(sum2, sq2, invCnt);

    // spikes from h = leaky(gn2(y2)): x_lr (axis H) -> bufC, x_td (axis W) -> bufD
    lif_kernel<<<24576, blk, 0, stream>>>(bufB, sum2, sq2, g2, be2, tau1, vth1, bufC, 0);
    lif_kernel<<<24576, blk, 0, stream>>>(bufB, sum2, sq2, g2, be2, tau2, vth2, bufD, 1);

    // y4 = conv1x1(x_lr, w21); stats4.  y5 = conv1x1(x_td, w22); stats5.
    conv1x1_wmma<0, 1><<<4096, blk, 0, stream>>>(
        bufC, w21, b21, bufA, nullptr, nullptr, nullptr, nullptr, sum4, sq4);
    conv1x1_wmma<0, 1><<<4096, blk, 0, stream>>>(
        bufD, w22, b22, bufB, nullptr, nullptr, nullptr, nullptr, sum5, sq5);
    reduce_stats<<<1, 192, 0, stream>>>(sum4, sq4, invCnt);
    reduce_stats<<<1, 192, 0, stream>>>(sum5, sq5, invCnt);

    // s_pre = leaky(gn4(y4)) + leaky(gn5(y5)); stats3
    add_gn<<<Bsz * DIM * 16, blk, 0, stream>>>(
        bufA, bufB, sum4, sq4, g4, be4, sum5, sq5, g5, be5, bufC, sum3, sq3);
    reduce_stats<<<1, 192, 0, stream>>>(sum3, sq3, invCnt);

    // out = conv1x1(gn3(s_pre), w3, b3)
    conv1x1_wmma<1, 0><<<4096, blk, 0, stream>>>(
        bufC, w3, b3, (float*)d_out, sum3, sq3, g3, be3, nullptr, nullptr);
}
